// VoxelNetBackbone_28097676050524
// MI455X (gfx1250) — matrix operand order repaired and checked by reading, not compiled, not run on hardware
//
#include <hip/hip_runtime.h>
#include <hip/hip_bf16.h>
#include <stdint.h>

// ---------------------------------------------------------------------------
// VoxelNet sparse backbone for gfx1250 (MI455X), wave32 + WMMA f16.
//
// Key layout decision: activations are stored channel-last f16 in grids with
// a 1-voxel ZERO HALO (66^3 per batch). Every 3^3-stencil load in both conv
// kernels is then in-bounds by construction -> the WMMA inner loops contain
// no predicates and no branches, just base + immediate-offset b128 loads.
//
// Pipeline (all on `stream`, all scratch in d_ws):
//  0. init    zero BN stats + out
//  1. hzero   zero the halo of padded h
//  2. prep    mask0 = (occ==0); xh = padded masked voxels [B][66^3][4] f16
//  3. packw1 / packw2: weights -> exact per-lane WMMA B layouts
//  4. mask1   3^3 dilation of mask0 + active count
//  5. conv1   16 voxels x 16 cout per wave, 4x v_wmma_f32_16x16x32_f16
//  6. bnparams1 / 7. bnapply1 (in-place on padded h interior)
//  8. mask2   stride-2 dilation of m1
//  9. conv2   16 voxels x 16 cout per wave, 27 taps x 2 K-chunks = 54 WMMA,
//             fully unrolled, all loads immediate-offset (halo-padded h)
// 10. bnparams2
// 11. final   masked BN+ReLU on h2 + mean over 32^3 -> out[4][128]
//
// Padded h (f16, 147 MB) fits in the 192 MB L2, so conv2's 27-tap re-reads
// stay on-chip; limiters are conv2 WMMA throughput and L2 bandwidth.
// ---------------------------------------------------------------------------

typedef __attribute__((ext_vector_type(16))) _Float16 v16h;
typedef __attribute__((ext_vector_type(8)))  float    v8f;

union Frag16 {
    v16h     v;
    _Float16 h[16];
    uint32_t u32[8];
    uint2    u64[4];
    uint4    u128[2];
};

#define BATCH 4
#define D3    262144          // 64^3
#define DO3   32768           // 32^3
#define PD    66              // padded dim
#define PD2   4356            // 66^2
#define PD3   287496          // 66^3
#define NPAD  (BATCH * PD3)   // padded voxels

// ---------------------------------------------------------------- init ------
__global__ void vx_init_kernel(float* stats1, float* stats2, float* out) {
    int t = threadIdx.x;
    if (t < 129) stats1[t] = 0.0f;   // sum[64], sumsq[64], count
    if (t < 257) stats2[t] = 0.0f;   // sum[128], sumsq[128], count
    if (t < 512) out[t]    = 0.0f;   // [4][128]
}

// ---------------------------------------------------------------- hzero -----
// Zero the halo voxels of padded h (64 channels each). Interior untouched.
__global__ void vx_hzero_kernel(_Float16* __restrict__ h) {
    int i = blockIdx.x * 256 + threadIdx.x;
    if (i >= NPAD) return;
    int s  = i % PD3;
    int pz = s / PD2;
    int r  = s % PD2;
    int py = r / PD;
    int px = r % PD;
    bool interior = (pz >= 1 && pz <= 64 && py >= 1 && py <= 64 &&
                     px >= 1 && px <= 64);
    if (!interior) {
        uint4  z4 = make_uint4(0u, 0u, 0u, 0u);
        uint4* p  = (uint4*)(h + (size_t)i * 64);
#pragma unroll
        for (int j = 0; j < 8; ++j) p[j] = z4;
    }
}

// ---------------------------------------------------------------- prep ------
// Full padded write: interior = masked voxels (f16 channel-last), halo = 0.
__global__ void vx_prep_kernel(const float* __restrict__ vox,
                               const int*   __restrict__ occ,
                               _Float16* __restrict__ xh,
                               uint8_t*  __restrict__ mask0) {
    int i = blockIdx.x * 256 + threadIdx.x;
    if (i >= NPAD) return;
    int b  = i / PD3;
    int s  = i % PD3;
    int pz = s / PD2;
    int r  = s % PD2;
    int py = r / PD;
    int px = r % PD;
    size_t xo = (size_t)i * 4;
    if (pz >= 1 && pz <= 64 && py >= 1 && py <= 64 && px >= 1 && px <= 64) {
        int z = pz - 1, y = py - 1, x = px - 1;
        int us = (z << 12) | (y << 6) | x;
        int ui = b * D3 + us;
        uint8_t m = (occ[ui] == 0) ? 1 : 0;
        mask0[ui] = m;
        float mf = (float)m;
#pragma unroll
        for (int c = 0; c < 4; ++c)
            xh[xo + c] = (_Float16)(vox[((size_t)b * 4 + c) * D3 + us] * mf);
    } else {
#pragma unroll
        for (int c = 0; c < 4; ++c) xh[xo + c] = (_Float16)0.0f;
    }
}

// --------------------------------------------------------------- packw1 -----
__global__ void vx_packw1_kernel(const float* __restrict__ w1,
                                 _Float16* __restrict__ wpack1) {
    int t    = blockIdx.x * 256 + threadIdx.x;   // t < 8192 (grid exact)
    int i    = t & 15;
    int lane = (t >> 4) & 31;
    int cog  = (t >> 9) & 3;
    int ch   = (t >> 11) & 3;
    int co   = cog * 16 + (lane & 15);
    int ktot = ch * 32 + (i & 8) * 2 + (lane >> 4) * 8 + (i & 7);
    int tap  = ktot >> 2;
    int ci   = ktot & 3;
    float v  = (tap < 27) ? w1[((size_t)co * 4 + ci) * 27 + tap] : 0.0f;
    wpack1[t] = (_Float16)v;
}

// --------------------------------------------------------------- packw2 -----
__global__ void vx_packw2_kernel(const float* __restrict__ w2,
                                 _Float16* __restrict__ wpack2) {
    int t    = blockIdx.x * 256 + threadIdx.x;   // t < 221184 (grid exact)
    int i    = t & 15;
    int lane = (t >> 4) & 31;
    int cog  = (t >> 9) & 7;
    int ch   = (t >> 12) & 1;
    int tap  = t >> 13;
    int co   = cog * 16 + (lane & 15);
    int ci   = ch * 32 + (i & 8) * 2 + (lane >> 4) * 8 + (i & 7);
    wpack2[t] = (_Float16)w2[((size_t)co * 64 + ci) * 27 + tap];
}

// ---------------------------------------------------------------- mask1 -----
__global__ void vx_mask1_kernel(const uint8_t* __restrict__ mask0,
                                uint8_t* __restrict__ m1,
                                float* __restrict__ cnt) {
    int i = blockIdx.x * 256 + threadIdx.x;      // grid exact: BATCH*D3
    int b = i >> 18;
    int s = i & (D3 - 1);
    int z = s >> 12, y = (s >> 6) & 63, x = s & 63;
    int any = 0;
    for (int kz = -1; kz <= 1; ++kz)
        for (int ky = -1; ky <= 1; ++ky)
            for (int kx = -1; kx <= 1; ++kx) {
                int iz = z + kz, iy = y + ky, ix = x + kx;
                if ((unsigned)iz < 64u && (unsigned)iy < 64u && (unsigned)ix < 64u)
                    any |= mask0[(((b * 64 + iz) * 64 + iy) * 64) + ix];
            }
    m1[i] = (uint8_t)any;
    unsigned long long bal = __ballot(any != 0);
    if ((threadIdx.x & 31) == 0)
        atomicAdd(cnt, (float)__popcll(bal));
}

// ---------------------------------------------------------------- conv1 -----
// One wave = 16 consecutive-x voxels x 16 output channels.
// K = tap*4+ci (padded to 128) -> 4 WMMA 16x16x32 f16 per tile.
// xh is halo-padded: ALL loads are unconditional. Per-lane tap (t0 for lanes
// 0-15, t0+2 for 16-31) needs at most one cndmask of two constant offsets.
// Pad taps (>=27) have zero B columns, so their A data is irrelevant (finite).
__global__ void vx_conv1_kernel(const _Float16* __restrict__ xh,
                                const _Float16* __restrict__ wpack1,
                                const uint8_t* __restrict__ m1,
                                const float* __restrict__ b1,
                                _Float16* __restrict__ h,
                                float* __restrict__ stats1) {
    int wave = blockIdx.x * 8 + (threadIdx.x >> 5);
    int lane = threadIdx.x & 31;
    int cog  = wave & 3;
    int tile = wave >> 2;                  // tiles = 4*64*64*4 = 65536
    int xb   = (tile & 3) * 16;
    int y    = (tile >> 2) & 63;
    int z    = (tile >> 8) & 63;
    int b    = tile >> 14;
    int g    = lane >> 4;
    int xcoord = xb + (lane & 15);         // A-row voxel for this lane

    // Padded coord of tap (0,0,0) input voxel (z-1,y-1,x-1) is (z,y,x).
    const _Float16* xbase = xh +
        ((size_t)(((b * PD + z) * PD + y) * PD + xcoord)) * 4;
    const _Float16* wbase = wpack1 + ((size_t)(cog * 32 + lane)) * 16;

    v8f acc = {0.f, 0.f, 0.f, 0.f, 0.f, 0.f, 0.f, 0.f};

#pragma unroll
    for (int ch = 0; ch < 4; ++ch) {
        Frag16 a, bm;
        bm.u128[0] = *(const uint4*)(wbase + ch * 2048);
        bm.u128[1] = *(const uint4*)(wbase + ch * 2048 + 8);
#pragma unroll
        for (int gi = 0; gi < 2; ++gi) {
#pragma unroll
            for (int j = 0; j < 2; ++j) {
                const int t0 = ch * 8 + gi * 4 + j;      // lanes 0-15
                const int t1 = t0 + 2;                   // lanes 16-31
                const bool v0 = (t0 < 27), v1 = (t1 < 27);
                const int kz0 = t0 / 9, ky0 = (t0 % 9) / 3, kx0 = t0 % 3;
                const int kz1 = t1 / 9, ky1 = (t1 % 9) / 3, kx1 = t1 % 3;
                const int off0 = (kz0 * PD2 + ky0 * PD + kx0) * 4;
                const int off1 = (kz1 * PD2 + ky1 * PD + kx1) * 4;
                int off;
                if (v0 && v1)      off = g ? off1 : off0;  // one cndmask
                else if (v0)       off = off0;             // pad tap: B cols 0
                else               off = 0;                // both pad: B cols 0
                a.u64[gi * 2 + j] = *(const uint2*)(xbase + off);
            }
        }
        acc = __builtin_amdgcn_wmma_f32_16x16x32_f16(
            false, a.v, false, bm.v, (short)0, acc, false, false);
    }

    // Epilogue: bias, mask, store pre-BN h (padded, channel-last), BN1 stats.
    int   co    = cog * 16 + (lane & 15);
    float bias  = b1[co];
    float s1 = 0.f, s2 = 0.f;
    int   mbase = (lane >> 4) * 8;         // C rows: m = (lane/16)*8 + r
#pragma unroll
    for (int r = 0; r < 8; ++r) {
        int    xo  = xb + mbase + r;
        size_t ui  = (((size_t)b * 64 + z) * 64 + y) * 64 + xo;       // mask
        size_t pvi = (((size_t)b * PD + z + 1) * PD + y + 1) * PD + (xo + 1);
        float  val = acc[r] + bias;
        if (!m1[ui]) val = 0.f;
        h[pvi * 64 + co] = (_Float16)val;
        s1 += val;
        s2 += val * val;
    }
    s1 += __shfl_xor(s1, 16);
    s2 += __shfl_xor(s2, 16);
    if (lane < 16) {
        atomicAdd(&stats1[co], s1);
        atomicAdd(&stats1[64 + co], s2);
    }
}

// -------------------------------------------------------------- bnparams ----
__global__ void vx_bnparams_kernel(const float* __restrict__ stats,
                                   const float* __restrict__ gamma,
                                   const float* __restrict__ beta,
                                   float* __restrict__ bnp, int C) {
    int c = threadIdx.x;
    if (c >= C) return;
    float cnt  = fmaxf(stats[2 * C], 1.0f);
    float mean = stats[c] / cnt;
    float var  = stats[C + c] / cnt - mean * mean;
    float sc   = gamma[c] * rsqrtf(var + 1e-5f);
    bnp[c]     = sc;
    bnp[C + c] = beta[c] - mean * sc;
}

// -------------------------------------------------------------- bnapply1 ----
// In-place BN+ReLU on the interior of padded h.
__global__ void vx_bnapply1_kernel(_Float16* __restrict__ h,
                                   const uint8_t* __restrict__ m1,
                                   const float* __restrict__ bnp) {
    int t  = blockIdx.x * 256 + threadIdx.x;   // grid exact: 4*D3*64/8
    int vi = t >> 3;                           // unpadded voxel id
    int c0 = (t & 7) * 8;
    int b  = vi >> 18;
    int s  = vi & (D3 - 1);
    int z  = s >> 12, y = (s >> 6) & 63, x = s & 63;
    size_t pvi = (((size_t)b * PD + z + 1) * PD + y + 1) * PD + (x + 1);
    _Float16* p = h + pvi * 64 + c0;
    union { uint4 u; _Float16 hh[8]; } d;
    d.u = *(const uint4*)p;
    if (m1[vi]) {
#pragma unroll
        for (int j = 0; j < 8; ++j) {
            float v = (float)d.hh[j];
            v = fmaxf(bnp[c0 + j] * v + bnp[64 + c0 + j], 0.0f);
            d.hh[j] = (_Float16)v;
        }
    } else {
        d.u = make_uint4(0u, 0u, 0u, 0u);
    }
    *(uint4*)p = d.u;
}

// ---------------------------------------------------------------- mask2 -----
__global__ void vx_mask2_kernel(const uint8_t* __restrict__ m1,
                                uint8_t* __restrict__ m2,
                                float* __restrict__ cnt) {
    int i = blockIdx.x * 256 + threadIdx.x;    // grid exact: BATCH*DO3
    int b = i >> 15;
    int s = i & (DO3 - 1);
    int z = s >> 10, y = (s >> 5) & 31, x = s & 31;
    int any = 0;
    for (int kz = 0; kz < 3; ++kz)
        for (int ky = 0; ky < 3; ++ky)
            for (int kx = 0; kx < 3; ++kx) {
                int iz = 2 * z - 1 + kz, iy = 2 * y - 1 + ky, ix = 2 * x - 1 + kx;
                if ((unsigned)iz < 64u && (unsigned)iy < 64u && (unsigned)ix < 64u)
                    any |= m1[(((b * 64 + iz) * 64 + iy) * 64) + ix];
            }
    m2[i] = (uint8_t)any;
    unsigned long long bal = __ballot(any != 0);
    if ((threadIdx.x & 31) == 0)
        atomicAdd(cnt, (float)__popcll(bal));
}

// ---------------------------------------------------------------- conv2 -----
// One wave = 16 output voxels x 16 output channels; 27 taps x 2 K-chunks
// = 54x v_wmma_f32_16x16x32_f16, fully unrolled. h is halo-padded, so the
// inner loop is branch- and predicate-free: per (tap,chunk) it is exactly
// 4 immediate-offset global_load_b128 + 1 WMMA off two fixed base addresses.
__global__ void vx_conv2_kernel(const _Float16* __restrict__ h,
                                const _Float16* __restrict__ wpack2,
                                const uint8_t* __restrict__ m2,
                                const float* __restrict__ b2,
                                float* __restrict__ h2,
                                float* __restrict__ stats2) {
    int wave = blockIdx.x * 8 + (threadIdx.x >> 5);
    int lane = threadIdx.x & 31;
    int cog  = wave & 7;
    int tile = wave >> 3;                  // tiles = 4*32*32*2 = 8192
    int xb   = (tile & 1) * 16;
    int y    = (tile >> 1) & 31;
    int z    = (tile >> 6) & 31;
    int b    = tile >> 11;
    int g    = lane >> 4;
    int xcoord = xb + (lane & 15);

    // Padded coord of tap (0,0,0) input voxel (2z-1,2y-1,2x-1) is (2z,2y,2x).
    const _Float16* habase = h +
        ((size_t)(((b * PD + 2 * z) * PD + 2 * y) * PD + 2 * xcoord)) * 64;
    const _Float16* wbase = wpack2 + ((size_t)(cog * 32 + lane)) * 16;

    v8f acc = {0.f, 0.f, 0.f, 0.f, 0.f, 0.f, 0.f, 0.f};

#pragma unroll
    for (int kz = 0; kz < 3; ++kz) {
#pragma unroll
        for (int ky = 0; ky < 3; ++ky) {
#pragma unroll
            for (int kx = 0; kx < 3; ++kx) {
                const int tap  = (kz * 3 + ky) * 3 + kx;
                const int aoff = (kz * PD2 + ky * PD + kx) * 64;   // halfs
#pragma unroll
                for (int ch = 0; ch < 2; ++ch) {
                    Frag16 a, bm;
                    const int woff = (tap * 2 + ch) * 4096;        // halfs
                    bm.u128[0] = *(const uint4*)(wbase + woff);
                    bm.u128[1] = *(const uint4*)(wbase + woff + 8);
                    const int cA = ch * 32 + g * 8;
                    a.u128[0] = *(const uint4*)(habase + aoff + cA);
                    a.u128[1] = *(const uint4*)(habase + aoff + cA + 16);
                    acc = __builtin_amdgcn_wmma_f32_16x16x32_f16(
                        false, a.v, false, bm.v, (short)0, acc, false, false);
                }
            }
        }
    }

    int   co    = cog * 16 + (lane & 15);
    float bias  = b2[co];
    float s1 = 0.f, s2 = 0.f;
    int   mbase = (lane >> 4) * 8;
#pragma unroll
    for (int r = 0; r < 8; ++r) {
        int    xo = xb + mbase + r;
        size_t vo = (((size_t)b * 32 + z) * 32 + y) * 32 + xo;
        float  val = acc[r] + bias;
        if (!m2[vo]) val = 0.f;
        h2[vo * 128 + co] = val;
        s1 += val;
        s2 += val * val;
    }
    s1 += __shfl_xor(s1, 16);
    s2 += __shfl_xor(s2, 16);
    if (lane < 16) {
        atomicAdd(&stats2[co], s1);
        atomicAdd(&stats2[128 + co], s2);
    }
}

// ---------------------------------------------------------------- final -----
__global__ void vx_final_kernel(const float* __restrict__ h2,
                                const uint8_t* __restrict__ m2,
                                const float* __restrict__ bnp2,
                                float* __restrict__ out) {
    int b     = blockIdx.x >> 5;           // 128 blocks, 128 threads (co)
    int chunk = blockIdx.x & 31;
    int co    = threadIdx.x;
    float sc = bnp2[co], sh = bnp2[128 + co];
    float acc = 0.0f;
    int vbase = chunk * 1024;
    for (int v = 0; v < 1024; ++v) {
        size_t vo = (size_t)b * DO3 + vbase + v;
        float val = h2[vo * 128 + co];
        val = m2[vo] ? fmaxf(sc * val + sh, 0.0f) : 0.0f;
        acc += val;
    }
    atomicAdd(&out[b * 128 + co], acc * (1.0f / 32768.0f));
}

// ------------------------------------------------------------- launcher -----
extern "C" void kernel_launch(void* const* d_in, const int* in_sizes, int n_in,
                              void* d_out, int out_size, void* d_ws,
                              size_t ws_size, hipStream_t stream) {
    (void)in_sizes; (void)n_in; (void)out_size; (void)ws_size;

    const float* vox = (const float*)d_in[0];
    const int*   occ = (const int*)d_in[1];
    const float* w1  = (const float*)d_in[2];
    const float* b1  = (const float*)d_in[3];
    const float* g1  = (const float*)d_in[4];
    const float* be1 = (const float*)d_in[5];
    const float* w2  = (const float*)d_in[6];
    const float* b2  = (const float*)d_in[7];
    const float* g2  = (const float*)d_in[8];
    const float* be2 = (const float*)d_in[9];
    float* out = (float*)d_out;

    char* base = (char*)d_ws;
    size_t off = 0;
    auto take = [&](size_t bytes) {
        size_t o = off;
        off = (off + bytes + 255) & ~(size_t)255;
        return o;
    };
    _Float16* xh     = (_Float16*)(base + take((size_t)NPAD * 4 * 2));
    uint8_t*  mask0  = (uint8_t*) (base + take((size_t)BATCH * D3));
    uint8_t*  m1     = (uint8_t*) (base + take((size_t)BATCH * D3));
    uint8_t*  m2     = (uint8_t*) (base + take((size_t)BATCH * DO3));
    _Float16* h      = (_Float16*)(base + take((size_t)NPAD * 64 * 2));
    float*    h2     = (float*)   (base + take((size_t)BATCH * DO3 * 128 * 4));
    _Float16* wpack1 = (_Float16*)(base + take((size_t)4 * 4 * 32 * 16 * 2));
    _Float16* wpack2 = (_Float16*)(base + take((size_t)27 * 2 * 8 * 32 * 16 * 2));
    float*    stats1 = (float*)   (base + take(129 * 4));
    float*    stats2 = (float*)   (base + take(257 * 4));
    float*    bnp1   = (float*)   (base + take(128 * 4));
    float*    bnp2   = (float*)   (base + take(256 * 4));

    int padBlocks = (NPAD + 255) / 256;

    vx_init_kernel<<<1, 1024, 0, stream>>>(stats1, stats2, out);
    vx_hzero_kernel<<<padBlocks, 256, 0, stream>>>(h);
    vx_prep_kernel<<<padBlocks, 256, 0, stream>>>(vox, occ, xh, mask0);
    vx_packw1_kernel<<<8192 / 256, 256, 0, stream>>>(w1, wpack1);
    vx_packw2_kernel<<<221184 / 256, 256, 0, stream>>>(w2, wpack2);
    vx_mask1_kernel<<<BATCH * D3 / 256, 256, 0, stream>>>(mask0, m1, &stats1[128]);
    // conv1: 65536 tiles * 4 cogroups = 262144 waves, 8 waves/block
    vx_conv1_kernel<<<262144 / 8, 256, 0, stream>>>(xh, wpack1, m1, b1, h, stats1);
    vx_bnparams_kernel<<<1, 64, 0, stream>>>(stats1, g1, be1, bnp1, 64);
    vx_bnapply1_kernel<<<(BATCH * (size_t)D3 * 64 / 8) / 256, 256, 0, stream>>>(h, m1, bnp1);
    vx_mask2_kernel<<<BATCH * DO3 / 256, 256, 0, stream>>>(m1, m2, &stats2[256]);
    // conv2: 8192 tiles * 8 cogroups = 65536 waves, 8 waves/block
    vx_conv2_kernel<<<65536 / 8, 256, 0, stream>>>(h, wpack2, m2, b2, h2, stats2);
    vx_bnparams_kernel<<<1, 128, 0, stream>>>(stats2, g2, be2, bnp2, 128);
    vx_final_kernel<<<128, 128, 0, stream>>>(h2, m2, bnp2, out);
}
